// LocationSensitiveAttention_47081431498882
// MI455X (gfx1250) — matrix operand order repaired and checked by reading, not compiled, not run on hardware
//
#include <hip/hip_runtime.h>
#include <hip/hip_bf16.h>
#include <stddef.h>

// ---------------------------------------------------------------------------
// LocationSensitiveAttention, fused for MI455X (gfx1250, wave32, WMMA bf16)
//
//   B=32, T=2048, D=512, C=32, K=31
//   e[b,t] = Wv . tanh( q[b] + memory[b,t]@Wm + locfeat[b,t]@Wloc + biases )
//   a = softmax(mask(e));  ctx = a @ memory
//
// Dominant cost: memory@Wm GEMM (M=B*T=65536, N=512, K=512) -> WMMA bf16.
// Wloc (K=32) is concatenated onto Wm (K=512) -> one K=544 GEMM.
// Epilogue (fast tanh on the TRANS pipe, dot with Wv) fused -> the [B,T,D]
// intermediate never hits HBM.  M-tile = 64 rows/block for 4x B-operand
// reuse out of L2 (~64 FLOP/L2-byte).
// ---------------------------------------------------------------------------

typedef __attribute__((ext_vector_type(16))) __bf16 v16bf;
typedef __attribute__((ext_vector_type(8)))  float  v8f;

#define BB   32
#define TT   2048
#define DD   512
#define CC   32
#define KW   31
#define KTOT 544                  // 512 (Wm) + 32 (Wloc)
#define KSTEPS 17                 // 544 / 32
#define MROWS ((size_t)BB * TT)   // 65536
#define MTILE 64                  // rows per block in the main GEMM
#define TCHUNKS 16                // T-split for the ctx pass

// Branch-free tanh on the transcendental pipe: v_exp_f32 + v_rcp_f32.
// tanh(x) = 1 - 2/(exp(2x)+1): exp->inf gives +1, exp->0 gives -1, no NaN.
__device__ __forceinline__ float fast_tanh(float x) {
    return 1.0f - __fdividef(2.0f, __expf(2.0f * x) + 1.0f);
}

// ---------------------------------------------------------------------------
// Kernel 1: pre-swizzle B = [Wm ; Wloc] (544 x 512 fp32) into bf16 WMMA
// B-fragment order:  flat = ((ks*32 + nt)*32 + lane)*16 + e
//   B frag (32x16 bf16): lanes 0-15 hold K={0..7,16..23} (col = lane&15),
//                        lanes 16-31 hold K={8..15,24..31}.  k_in_step = (e&7) + 8*hi + 16*(e>>3).
// ---------------------------------------------------------------------------
__global__ __launch_bounds__(256) void prep_b_kernel(
    const float* __restrict__ Wm, const float* __restrict__ Wloc,
    __bf16* __restrict__ Bswz)
{
    int idx = blockIdx.x * 256 + threadIdx.x;       // 0 .. 278527
    if (idx >= KSTEPS * 32 * 32 * 16) return;
    int e    = idx & 15;
    int lane = (idx >> 4) & 31;
    int nt   = (idx >> 9) & 31;
    int ks   = idx >> 14;
    int hi   = lane >> 4;
    int kr   = (e & 7) + 8 * hi + 16 * (e >> 3);     // B-matrix k within step
    int k    = ks * 32 + kr;
    int n    = nt * 16 + (lane & 15);
    float v  = (k < DD) ? Wm[(size_t)k * DD + n]
                        : Wloc[(size_t)(k - DD) * DD + n];
    Bswz[idx] = (__bf16)v;
}

// ---------------------------------------------------------------------------
// Kernel 2: qsum[b,n] = bq[n] + bm[n] + bloc[n] + sum_k query[b,k]*Wq[k,n]
// ---------------------------------------------------------------------------
__global__ __launch_bounds__(256) void qsum_kernel(
    const float* __restrict__ query, const float* __restrict__ Wq,
    const float* __restrict__ bq, const float* __restrict__ bm,
    const float* __restrict__ bloc, float* __restrict__ qsum)
{
    int b = blockIdx.x, tid = threadIdx.x;
    for (int n = tid; n < DD; n += 256) {
        float acc = bq[n] + bm[n] + bloc[n];
        for (int k = 0; k < DD; ++k)
            acc += query[(size_t)b * DD + k] * Wq[(size_t)k * DD + n];
        qsum[(size_t)b * DD + n] = acc;
    }
}

// ---------------------------------------------------------------------------
// Kernel 3: 1D conv (cross-correlation, pad=15) over [prev;cum] channels.
// locfeat[b,t,c] = bconv[c] + sum_k W[c,0,k]*prev[t+k-15] + W[c,1,k]*cum[...]
// ---------------------------------------------------------------------------
__global__ __launch_bounds__(256) void conv_kernel(
    const float* __restrict__ prev, const float* __restrict__ cum,
    const float* __restrict__ Wconv, const float* __restrict__ bconv,
    float* __restrict__ locfeat)
{
    __shared__ float sp[256 + 2 * (KW / 2)];
    __shared__ float sc[256 + 2 * (KW / 2)];
    __shared__ float sw[CC * 2 * KW];
    __shared__ float sb[CC];
    int b = blockIdx.y, tchunk = blockIdx.x, tid = threadIdx.x;
    int tbase = tchunk * 256;
    for (int i = tid; i < 256 + 30; i += 256) {
        int t = tbase + i - 15;
        bool ok = (t >= 0) && (t < TT);
        sp[i] = ok ? prev[(size_t)b * TT + t] : 0.f;
        sc[i] = ok ? cum [(size_t)b * TT + t] : 0.f;
    }
    for (int i = tid; i < CC * 2 * KW; i += 256) sw[i] = Wconv[i];
    if (tid < CC) sb[tid] = bconv[tid];
    __syncthreads();

    int t = tbase + tid;
    float* out = locfeat + ((size_t)b * TT + t) * CC;
    for (int c = 0; c < CC; ++c) {
        float acc = sb[c];
        const float* w = &sw[c * 2 * KW];
        #pragma unroll
        for (int k = 0; k < KW; ++k)
            acc += w[k] * sp[tid + k] + w[KW + k] * sc[tid + k];
        out[c] = acc;
    }
}

// ---------------------------------------------------------------------------
// Kernel 4 (workhorse): per block, 64 rows x 512 cols x 544 K WMMA GEMM
// with fused epilogue:  e[row] = bv + sum_n tanh(acc[n] + qsum[b,n]) * Wv[n]
//
// Block = 512 threads = 16 waves. Wave w owns N columns [w*32, w*32+32)
// (2 WMMA n-tiles) and loops 4 M-subtiles -> 8 WMMAs per wave per K-step
// against only 2 B-fragment loads (4x the B reuse of an M=16 tile).
// A tile staged fp32->bf16 into LDS in A-fragment order; B fragments stream
// pre-swizzled from L2 (557 KB, resident).
// A frag (16x32 bf16): lane<16 holds M=lane, k = {0..7,16..23};
// lane>=16 holds M=lane-16, k = {8..15,24..31}.  (m,k)->(l,e):
//   hi=(kr>>3)&1, l=m+16*hi, e=(kr&7)+8*(kr>>4)
// ---------------------------------------------------------------------------
__global__ __launch_bounds__(512) void attn_energy_gemm(
    const float* __restrict__ memory, const float* __restrict__ locfeat,
    const __bf16* __restrict__ Bswz, const float* __restrict__ qsum,
    const float* __restrict__ Wv, const float* __restrict__ bv,
    float* __restrict__ e_out)
{
    __shared__ __align__(32) __bf16 Asw[KSTEPS * 4 * 32 * 16];  // 68 KB
    __shared__ float e_acc[MTILE];

    const int tid  = threadIdx.x;
    const int lane = tid & 31;
    const int wave = tid >> 5;
    const size_t row0 = (size_t)blockIdx.x * MTILE;
    const int b = (int)(row0 >> 11);                 // row / T

    // ---- stage A tile (64 x 544), fp32 -> bf16, fragment-order in LDS ----
    for (int idx = tid; idx < MTILE * KTOT; idx += 512) {
        int m = idx / KTOT;
        int k = idx - m * KTOT;
        float v = (k < DD)
            ? memory [(row0 + m) * DD + k]
            : locfeat[(row0 + m) * CC + (k - DD)];
        int ms = m >> 4, ml = m & 15;
        int ks = k >> 5, kr = k & 31;
        int hi = (kr >> 3) & 1;
        int e  = (kr & 7) + ((kr >> 4) << 3);
        Asw[((ks * 4 + ms) * 32 + ml + 16 * hi) * 16 + e] = (__bf16)v;
    }
    if (tid < MTILE) e_acc[tid] = 0.f;
    __syncthreads();

    // ---- main WMMA loop: K = 544 in 17 steps of 32 ----
    v8f acc[4][2] = {};
    const int nt0 = wave * 2;
    for (int ks = 0; ks < KSTEPS; ++ks) {
        const __bf16* bp = Bswz + ((size_t)(ks * 32 + nt0) * 32 + lane) * 16;
        if (ks + 1 < KSTEPS)
            __builtin_prefetch(bp + (size_t)32 * 32 * 16, 0, 1);
        v16bf b0 = *(const v16bf*)(bp);
        v16bf b1 = *(const v16bf*)(bp + 32 * 16);
        #pragma unroll
        for (int ms = 0; ms < 4; ++ms) {
            v16bf a = *(const v16bf*)&Asw[((ks * 4 + ms) * 32 + lane) * 16];
            acc[ms][0] = __builtin_amdgcn_wmma_f32_16x16x32_bf16(
                false, a, false, b0, (short)0, acc[ms][0], false, false);
            acc[ms][1] = __builtin_amdgcn_wmma_f32_16x16x32_bf16(
                false, a, false, b1, (short)0, acc[ms][1], false, false);
        }
    }

    // ---- fused epilogue: fast tanh + dot with Wv, reduce over N via LDS ----
    // C/D layout: VGPR r, lane -> M = r + 8*(lane>>4), N = lane&15 (per tile)
    const int hi = lane >> 4, nl = lane & 15;
    #pragma unroll
    for (int ms = 0; ms < 4; ++ms) {
        float psum[8];
        #pragma unroll
        for (int r = 0; r < 8; ++r) psum[r] = 0.f;
        #pragma unroll
        for (int j = 0; j < 2; ++j) {
            int n = wave * 32 + j * 16 + nl;
            float qv = qsum[(size_t)b * DD + n];
            float wv = Wv[n];
            #pragma unroll
            for (int r = 0; r < 8; ++r)
                psum[r] += fast_tanh(acc[ms][j][r] + qv) * wv;
        }
        #pragma unroll
        for (int r = 0; r < 8; ++r)
            atomicAdd(&e_acc[ms * 16 + r + 8 * hi], psum[r]);  // ds_add_f32
    }
    __syncthreads();
    if (tid < MTILE)
        e_out[row0 + tid] = e_acc[tid] + bv[0];
}

// ---------------------------------------------------------------------------
// Kernel 5: masked softmax over T per batch row (32 blocks x 256 threads)
// ---------------------------------------------------------------------------
__global__ __launch_bounds__(256) void softmax_kernel(
    const float* __restrict__ e_ws, const unsigned char* __restrict__ mask,
    float* __restrict__ a_out)
{
    __shared__ float red[256];
    __shared__ int allm_s;
    int b = blockIdx.x, tid = threadIdx.x;
    float ev[8];
    int allm = 1;
    #pragma unroll
    for (int i = 0; i < 8; ++i) {
        int t = tid + i * 256;
        int mk = mask[(size_t)b * TT + t] ? 1 : 0;
        allm &= mk;
        ev[i] = mk ? -1e9f : e_ws[(size_t)b * TT + t];
    }
    if (tid == 0) allm_s = 1;
    __syncthreads();
    if (!allm) atomicAnd(&allm_s, 0);
    __syncthreads();
    if (allm_s && tid == 0) ev[0] = 0.f;           // reference all-masked fixup

    float m = ev[0];
    #pragma unroll
    for (int i = 1; i < 8; ++i) m = fmaxf(m, ev[i]);
    red[tid] = m; __syncthreads();
    for (int s = 128; s > 0; s >>= 1) {
        if (tid < s) red[tid] = fmaxf(red[tid], red[tid + s]);
        __syncthreads();
    }
    m = red[0]; __syncthreads();

    float ex[8]; float ssum = 0.f;
    #pragma unroll
    for (int i = 0; i < 8; ++i) { ex[i] = __expf(ev[i] - m); ssum += ex[i]; }
    red[tid] = ssum; __syncthreads();
    for (int s = 128; s > 0; s >>= 1) {
        if (tid < s) red[tid] += red[tid + s];
        __syncthreads();
    }
    float inv = 1.f / red[0];
    #pragma unroll
    for (int i = 0; i < 8; ++i)
        a_out[(size_t)b * TT + tid + i * 256] = ex[i] * inv;
}

// ---------------------------------------------------------------------------
// Kernel 6a: ctx partials over T-chunks (B x TCHUNKS blocks -> saturate HBM)
//   part[ch,b,d] = sum_{t in chunk ch} a[b,t] * memory[b,t,d]
// ---------------------------------------------------------------------------
__global__ __launch_bounds__(256) void ctx_partial_kernel(
    const float* __restrict__ a, const float* __restrict__ memory,
    float* __restrict__ part)
{
    int ch = blockIdx.x, b = blockIdx.y, tid = threadIdx.x;
    int d0 = tid * 2;
    int t0 = ch * (TT / TCHUNKS);
    const float* mrow = memory + ((size_t)b * TT + t0) * DD;
    const float* arow = a + (size_t)b * TT + t0;
    float ax = 0.f, ay = 0.f;
    for (int t = 0; t < TT / TCHUNKS; ++t) {
        float at = arow[t];
        float2 mv = *(const float2*)(mrow + (size_t)t * DD + d0);
        ax += at * mv.x; ay += at * mv.y;
    }
    float* p = part + ((size_t)ch * BB + b) * DD + d0;
    p[0] = ax; p[1] = ay;
}

// Kernel 6b: reduce partials -> ctx[b,d]
__global__ __launch_bounds__(256) void ctx_reduce_kernel(
    const float* __restrict__ part, float* __restrict__ ctx)
{
    int b = blockIdx.x, tid = threadIdx.x;
    int d0 = tid * 2;
    float ax = 0.f, ay = 0.f;
    for (int ch = 0; ch < TCHUNKS; ++ch) {
        const float* p = part + ((size_t)ch * BB + b) * DD + d0;
        ax += p[0]; ay += p[1];
    }
    ctx[(size_t)b * DD + d0]     = ax;
    ctx[(size_t)b * DD + d0 + 1] = ay;
}

// ---------------------------------------------------------------------------
extern "C" void kernel_launch(void* const* d_in, const int* in_sizes, int n_in,
                              void* d_out, int out_size, void* d_ws, size_t ws_size,
                              hipStream_t stream)
{
    const float* query     = (const float*)d_in[0];
    const float* memory    = (const float*)d_in[1];
    const float* prev_attn = (const float*)d_in[2];
    const float* cum_attn  = (const float*)d_in[3];
    const unsigned char* mask = (const unsigned char*)d_in[4];  // bool array
    const float* Wq    = (const float*)d_in[5];
    const float* bq    = (const float*)d_in[6];
    const float* Wm    = (const float*)d_in[7];
    const float* bm    = (const float*)d_in[8];
    const float* Wconv = (const float*)d_in[9];
    const float* bconv = (const float*)d_in[10];
    const float* Wloc  = (const float*)d_in[11];
    const float* bloc  = (const float*)d_in[12];
    const float* Wv    = (const float*)d_in[13];
    const float* bv    = (const float*)d_in[14];

    float* out     = (float*)d_out;
    float* ctx_out = out;                  // [B, D]  first in return order
    float* a_out   = out + BB * DD;        // [B, T]

    // workspace layout
    char* ws = (char*)d_ws;
    const size_t B_BYTES   = (size_t)KSTEPS * 32 * 32 * 16 * sizeof(__bf16); // 557056
    const size_t Q_BYTES   = (size_t)BB * DD * sizeof(float);                // 65536
    const size_t LOC_BYTES = MROWS * CC * sizeof(float);                     // 8 MiB
    const size_t E_BYTES   = MROWS * sizeof(float);                          // 256 KiB
    __bf16* Bswz    = (__bf16*)(ws);
    float*  qsum    = (float*)(ws + B_BYTES);
    float*  locfeat = (float*)(ws + B_BYTES + Q_BYTES);
    float*  e_ws    = (float*)(ws + B_BYTES + Q_BYTES + LOC_BYTES);
    float*  part    = (float*)(ws + B_BYTES + Q_BYTES + LOC_BYTES + E_BYTES);

    prep_b_kernel<<<(KSTEPS * 32 * 32 * 16 + 255) / 256, 256, 0, stream>>>(Wm, Wloc, Bswz);
    qsum_kernel<<<BB, 256, 0, stream>>>(query, Wq, bq, bm, bloc, qsum);
    conv_kernel<<<dim3(TT / 256, BB), 256, 0, stream>>>(prev_attn, cum_attn, Wconv, bconv, locfeat);
    attn_energy_gemm<<<(int)(MROWS / MTILE), 512, 0, stream>>>(memory, locfeat, Bswz, qsum, Wv, bv, e_ws);
    softmax_kernel<<<BB, 256, 0, stream>>>(e_ws, mask, a_out);
    ctx_partial_kernel<<<dim3(TCHUNKS, BB), 256, 0, stream>>>(a_out, memory, part);
    ctx_reduce_kernel<<<BB, 256, 0, stream>>>(part, ctx_out);
}
